// SelfAttention_49323404427787
// MI455X (gfx1250) — matrix operand order repaired and checked by reading, not compiled, not run on hardware
//
#include <hip/hip_runtime.h>

// ---------------------------------------------------------------------------
// Self-attention, bf16 WMMA pipeline for gfx1250 (wave32).
//   x:[4,4096,1024] f32, W_{q,k,v}:[1024,1024] f32  ->  out:[4,4096,1024] f32
// Stage 1: W -> bf16 transposed [N,K]
// Stage 2: QKV GEMM, fully unrolled K loop (no loop-carried fragments ->
//          no WMMA WAR-hazard copies)
// Stage 3: flash attention; K blocks DMA'd to LDS by the Tensor Data Mover
//          (TENSOR_LOAD_TO_LDS + s_wait_tensorcnt, async-LDS fallback),
//          S and P.V via WMMA, online softmax.
// ---------------------------------------------------------------------------

typedef __bf16 bf16_t;
typedef __attribute__((ext_vector_type(16))) __bf16 v16bf;
typedef __attribute__((ext_vector_type(8)))  __bf16 v8bf;
typedef __attribute__((ext_vector_type(8)))  float  v8f;
typedef unsigned int u32x4 __attribute__((ext_vector_type(4)));
typedef int          i32x4 __attribute__((ext_vector_type(4)));
typedef int          i32x8 __attribute__((ext_vector_type(8)));

#define D_MODEL 1024
#define SEQ     4096
#define BATCH   4
#define M_TOT   (BATCH * SEQ)

#if defined(__has_builtin)
#if __has_builtin(__builtin_amdgcn_tensor_load_to_lds)
#define HAVE_TDM 1
#endif
#endif
#ifndef HAVE_TDM
#define HAVE_TDM 0
#endif

__device__ __forceinline__ v16bf frag_combine(v8bf lo, v8bf hi) {
  return __builtin_shufflevector(lo, hi, 0,1,2,3,4,5,6,7,8,9,10,11,12,13,14,15);
}

// A fragment (16x32 bf16), row-major source, base -> (tileRow0, k0).
__device__ __forceinline__ v16bf load_frag_a_bf16(const bf16_t* base, int ld, int lane) {
  const int row = lane & 15;
  const int kb  = (lane >> 4) << 4;
  const bf16_t* p = base + (size_t)row * ld + kb;
  v8bf lo = *(const v8bf*)(p);
  v8bf hi = *(const v8bf*)(p + 8);
  return frag_combine(lo, hi);
}

// A fragment from f32 row-major source, converting to bf16 inline.
__device__ __forceinline__ v16bf load_frag_a_f32(const float* base, int ld, int lane) {
  const int row = lane & 15;
  const int kb  = (lane >> 4) << 4;
  const float* p = base + (size_t)row * ld + kb;
  v16bf a;
#pragma unroll
  for (int i = 0; i < 8; ++i) a[i]     = (bf16_t)p[i];
#pragma unroll
  for (int i = 0; i < 8; ++i) a[8 + i] = (bf16_t)p[8 + i];
  return a;
}

// B fragment (32x16 bf16). Source stores element (k, n) at n*ld + k.
__device__ __forceinline__ v16bf load_frag_b(const bf16_t* base, long ld, int lane) {
  const int col = lane & 15;
  const int kb  = (lane >> 4) << 4;
  const bf16_t* p = base + (size_t)col * ld + kb;
  v8bf lo = *(const v8bf*)(p);
  v8bf hi = *(const v8bf*)(p + 8);
  return frag_combine(lo, hi);
}

__device__ __forceinline__ v8f wmma_bf16(v16bf a, v16bf b, v8f c) {
  return __builtin_amdgcn_wmma_f32_16x16x32_bf16(false, a, false, b, (short)0, c,
                                                 false, false);
}

// ---- staging sync --------------------------------------------------------
__device__ __forceinline__ void wait_async0() {
#if __has_builtin(__builtin_amdgcn_s_wait_asynccnt)
  __builtin_amdgcn_s_wait_asynccnt(0);
#else
  asm volatile("s_wait_asynccnt 0" ::: "memory");
#endif
}
__device__ __forceinline__ void wait_tensor0() {
#if __has_builtin(__builtin_amdgcn_s_wait_tensorcnt)
  __builtin_amdgcn_s_wait_tensorcnt((short)0);
#else
  asm volatile("s_wait_tensorcnt 0" ::: "memory");
#endif
}
__device__ __forceinline__ void async_copy_b128(void* lds_dst, const void* gsrc) {
  unsigned loff = (unsigned)(size_t)lds_dst;   // LDS aperture: addr[31:0] = offset
  asm volatile("global_load_async_to_lds_b128 %0, %1, off"
               :: "v"(loff), "v"(gsrc) : "memory");
}

#if HAVE_TDM
// TDM: DMA a (512 feat x 32 key) bf16 tile into LDS with 16B pad per row
// (pad_interval = 256 dwords, pad_amount = 4 dwords -> LDS row = 1040 B).
__device__ __forceinline__ void tdm_load_tile(unsigned lds_off, const void* gaddr) {
  unsigned long long ga = (unsigned long long)(size_t)gaddr;
  u32x4 g0;
  g0[0] = 1u;                                            // count=1, user D#
  g0[1] = lds_off;                                       // lds_addr
  g0[2] = (unsigned)(ga & 0xffffffffull);                // global_addr[31:0]
  g0[3] = (unsigned)((ga >> 32) & 0x1ffffffull) | (2u << 30); // [56:32] | type=2
  i32x8 g1;
  g1[0] = (1 << 16)        // data_size = 2 bytes
        | (1 << 20)        // pad_enable
        | (7 << 22)        // pad_interval = 256 dwords (one 512-elem row)
        | (3 << 25);       // pad_amount = 4 dwords (16 B)
  g1[1] = (512 << 16);     // tensor_dim0 = 512 (low 16 bits into [31:16])
  g1[2] = (int)((unsigned)SEQ << 16);  // tensor_dim1 = 4096 rows -> [31:16]
  g1[3] = (512 << 16);     // tile_dim0 = 512 -> bits[127:112]
  g1[4] = 32;              // tile_dim1 = 32 keys
  g1[5] = D_MODEL;         // tensor_dim0_stride = 1024 elements
  g1[6] = 0;
  g1[7] = 0;
  i32x4 gz = {0, 0, 0, 0};
#if __clang_major__ >= 23
  i32x8 gz8 = {0, 0, 0, 0, 0, 0, 0, 0};
  __builtin_amdgcn_tensor_load_to_lds(g0, g1, gz, gz, gz8, 0);
#else
  __builtin_amdgcn_tensor_load_to_lds(g0, g1, gz, gz, 0);
#endif
}
#endif

// ---------------------------------------------------------------------------
// Stage 1: WT[n*1024 + k] = bf16( W[k*1024 + n] )
// ---------------------------------------------------------------------------
__global__ void transpose_w_kernel(const float* __restrict__ W,
                                   bf16_t* __restrict__ WT) {
  int i = blockIdx.x * blockDim.x + threadIdx.x;
  if (i >= D_MODEL * D_MODEL) return;
  int n = i >> 10;
  int k = i & (D_MODEL - 1);
  WT[i] = (bf16_t)W[(size_t)k * D_MODEL + n];
}

// ---------------------------------------------------------------------------
// Stage 2: QKV GEMM. Wave tile 16(M) x 64(N), K loop fully unrolled.
// ---------------------------------------------------------------------------
__global__ __launch_bounds__(256) void qkv_gemm_kernel(
    const float*  __restrict__ x,
    const bf16_t* __restrict__ WqT,
    const bf16_t* __restrict__ WkT,
    const bf16_t* __restrict__ WvT,
    bf16_t* __restrict__ Qh,
    bf16_t* __restrict__ Kh,
    bf16_t* __restrict__ Vt)
{
  const int lane  = threadIdx.x & 31;
  const int wave  = threadIdx.x >> 5;
  const int m0    = blockIdx.x * 128 + wave * 16;
  const int n0    = blockIdx.y * 64;
  const int which = blockIdx.z;
  const bf16_t* WT = (which == 0) ? WqT : (which == 1) ? WkT : WvT;
  const float* Xbase = x + (size_t)m0 * D_MODEL;

  v8f acc[4] = {};

#pragma unroll
  for (int k0 = 0; k0 < D_MODEL; k0 += 32) {
    v16bf a = load_frag_a_f32(Xbase + k0, D_MODEL, lane);
    v16bf bb[4];
#pragma unroll
    for (int nt = 0; nt < 4; ++nt)
      bb[nt] = load_frag_b(WT + (size_t)(n0 + nt * 16) * D_MODEL + k0,
                           D_MODEL, lane);
#pragma unroll
    for (int nt = 0; nt < 4; ++nt)
      acc[nt] = wmma_bf16(a, bb[nt], acc[nt]);
  }

  const int cl  = lane & 15;
  const int hi8 = (lane >> 4) * 8;
  if (which == 2) {
#pragma unroll
    for (int nt = 0; nt < 4; ++nt)
#pragma unroll
      for (int v = 0; v < 8; ++v)
        Vt[(size_t)(n0 + nt * 16 + cl) * M_TOT + (m0 + v + hi8)] = (bf16_t)acc[nt][v];
  } else {
    bf16_t* out = (which == 0) ? Qh : Kh;
#pragma unroll
    for (int nt = 0; nt < 4; ++nt)
#pragma unroll
      for (int v = 0; v < 8; ++v)
        out[(size_t)(m0 + v + hi8) * D_MODEL + (n0 + nt * 16 + cl)] = (bf16_t)acc[nt][v];
  }
}

// ---------------------------------------------------------------------------
// Stage 3: flash attention. Block = 64 queries, 256 threads, key step 32.
// K block in LDS as [2 half][32 key][520 bf16]  (row 1040 B: 260 words
// == 4 mod 64 banks -> conflict-free 16B-aligned column reads).
// ---------------------------------------------------------------------------
#define QB   64
#define KB   32
#define KROW 520
#define SM_SCALE 0.03125f

// B frag from padded K tile: element (feat f, key) ; f never straddles a half.
__device__ __forceinline__ v16bf load_frag_b_sk(const bf16_t* skbase, int key0,
                                                int f0, int lane) {
  const int key = key0 + (lane & 15);
  const int f   = f0 + ((lane >> 4) << 4);
  const int half = f >> 9;
  const int fl   = f & 511;
  const bf16_t* p = skbase + ((size_t)half * KB + key) * KROW + fl;
  v8bf lo = *(const v8bf*)(p);
  v8bf hi = *(const v8bf*)(p + 8);
  return frag_combine(lo, hi);
}

__global__ __launch_bounds__(256) void attn_kernel(
    const bf16_t* __restrict__ Qh,
    const bf16_t* __restrict__ Kh,
    const bf16_t* __restrict__ Vt,
    float* __restrict__ out)
{
  __shared__ __align__(16) bf16_t sK[2][2][KB][KROW];   // 2 bufs x 65 KB
  __shared__ __align__(16) float  sS[QB][KB + 1];
  __shared__ __align__(16) bf16_t sP[QB][40];
  __shared__ float sM[QB], sL[QB], sR[QB];

  const int lane = threadIdx.x & 31;
  const int wave = threadIdx.x >> 5;
  const int b    = blockIdx.y;
  const int q0   = blockIdx.x * QB;
  const size_t rowbase = (size_t)b * SEQ + q0;

  if (threadIdx.x < QB) { sM[threadIdx.x] = -1e30f; sL[threadIdx.x] = 0.0f; }

  v8f o[4][8] = {};

  const int smt = wave >> 1;
  const int snt = wave & 1;
  const int cl  = lane & 15;
  const int hi8 = (lane >> 4) * 8;

  // stage one 32-key K block into sK[bufSel] (TDM if available, async otherwise)
  auto stage_k = [&](int bufSel, int j0) {
    const bf16_t* src = Kh + ((size_t)b * SEQ + j0) * D_MODEL;
#if HAVE_TDM
    if (wave == 0) {
#pragma unroll
      for (int h = 0; h < 2; ++h)
        tdm_load_tile((unsigned)(size_t)&sK[bufSel][h][0][0], src + h * 512);
    }
#else
    for (int c = threadIdx.x; c < 2 * KB * 64; c += 256) {
      int half = c >> 11;
      int rem  = c & 2047;
      int row  = rem >> 6;        // 64 x 16B payload chunks per row
      int seg  = rem & 63;
      async_copy_b128(&sK[bufSel][half][row][seg * 8],
                      src + (size_t)row * D_MODEL + half * 512 + seg * 8);
    }
#endif
  };
  auto wait_stage = [&]() {
#if HAVE_TDM
    wait_tensor0();
#else
    wait_async0();
#endif
  };

  stage_k(0, 0);
  int buf = 0;

  for (int j0 = 0; j0 < SEQ; j0 += KB) {
    wait_stage();
    __syncthreads();
    if (j0 + KB < SEQ) stage_k(buf ^ 1, j0 + KB);

    // ---- Phase A: S = Q . K^T, fully unrolled K loop
    v8f s = {};
    const bf16_t* Abase = Qh + (rowbase + (size_t)smt * 16) * D_MODEL;
    const bf16_t* Kt = &sK[buf][0][0][0];
    const int key0 = snt * 16;
#pragma unroll
    for (int k0 = 0; k0 < D_MODEL; k0 += 32) {
      v16bf a  = load_frag_a_bf16(Abase + k0, D_MODEL, lane);
      v16bf kk = load_frag_b_sk(Kt, key0, k0, lane);
      s = wmma_bf16(a, kk, s);
    }
#pragma unroll
    for (int v = 0; v < 8; ++v)
      sS[smt * 16 + v + hi8][snt * 16 + cl] = s[v];
    __syncthreads();

    // ---- Phase B: online softmax (64 threads, one row each)
    if (threadIdx.x < QB) {
      const int r = threadIdx.x;
      float mr = -1e30f;
#pragma unroll
      for (int c = 0; c < KB; ++c)
        mr = fmaxf(mr, sS[r][c] * SM_SCALE);
      const float m_old = sM[r];
      const float m_new = fmaxf(m_old, mr);
      const float corr  = __expf(m_old - m_new);
      float sum = 0.0f;
#pragma unroll
      for (int c = 0; c < KB; ++c) {
        float e = __expf(sS[r][c] * SM_SCALE - m_new);
        sP[r][c] = (bf16_t)e;
        sum += e;
      }
      sM[r] = m_new;
      sL[r] = sL[r] * corr + sum;
      sR[r] = corr;
    }
    __syncthreads();

    // ---- Phase C: rescale O, then O += P . V (Vt frags, unrolled)
#pragma unroll
    for (int mt = 0; mt < 4; ++mt)
#pragma unroll
      for (int v = 0; v < 8; ++v) {
        const float f = sR[mt * 16 + v + hi8];
#pragma unroll
        for (int nt = 0; nt < 8; ++nt) o[mt][nt][v] *= f;
      }

    v16bf pfrag[4];
#pragma unroll
    for (int mt = 0; mt < 4; ++mt)
      pfrag[mt] = load_frag_a_bf16(&sP[mt * 16][0], 40, lane);

    const bf16_t* Vbase = Vt + (size_t)(wave * 128) * M_TOT + (size_t)b * SEQ + j0;
#pragma unroll
    for (int nt = 0; nt < 8; ++nt) {
      v16bf vb = load_frag_b(Vbase + (size_t)nt * 16 * M_TOT, M_TOT, lane);
#pragma unroll
      for (int mt = 0; mt < 4; ++mt)
        o[mt][nt] = wmma_bf16(pfrag[mt], vb, o[mt][nt]);
    }
    buf ^= 1;
  }

  // ---- Epilogue: O /= l, store f32 row-major
#pragma unroll
  for (int mt = 0; mt < 4; ++mt)
#pragma unroll
    for (int v = 0; v < 8; ++v) {
      const int row = mt * 16 + v + hi8;
      const float inv = 1.0f / sL[row];
      const size_t m = rowbase + row;
#pragma unroll
      for (int nt = 0; nt < 8; ++nt) {
        const int n = wave * 128 + nt * 16 + cl;
        out[m * D_MODEL + n] = o[mt][nt][v] * inv;
      }
    }
}

// ---------------------------------------------------------------------------
extern "C" void kernel_launch(void* const* d_in, const int* in_sizes, int n_in,
                              void* d_out, int out_size, void* d_ws, size_t ws_size,
                              hipStream_t stream) {
  const float* x  = (const float*)d_in[0];
  const float* Wq = (const float*)d_in[1];
  const float* Wk = (const float*)d_in[2];
  const float* Wv = (const float*)d_in[3];
  float* out = (float*)d_out;

  char* ws = (char*)d_ws;
  size_t off = 0;
  auto alloc = [&](size_t bytes) -> void* {
    void* p = ws + off;
    off = (off + bytes + 255) & ~(size_t)255;
    return p;
  };
  bf16_t* WqT = (bf16_t*)alloc((size_t)D_MODEL * D_MODEL * sizeof(bf16_t));
  bf16_t* WkT = (bf16_t*)alloc((size_t)D_MODEL * D_MODEL * sizeof(bf16_t));
  bf16_t* WvT = (bf16_t*)alloc((size_t)D_MODEL * D_MODEL * sizeof(bf16_t));
  bf16_t* Qh  = (bf16_t*)alloc((size_t)M_TOT * D_MODEL * sizeof(bf16_t));
  bf16_t* Kh  = (bf16_t*)alloc((size_t)M_TOT * D_MODEL * sizeof(bf16_t));
  bf16_t* Vt  = (bf16_t*)alloc((size_t)M_TOT * D_MODEL * sizeof(bf16_t));

  {
    dim3 grid((D_MODEL * D_MODEL + 255) / 256);
    transpose_w_kernel<<<grid, 256, 0, stream>>>(Wq, WqT);
    transpose_w_kernel<<<grid, 256, 0, stream>>>(Wk, WkT);
    transpose_w_kernel<<<grid, 256, 0, stream>>>(Wv, WvT);
  }
  {
    dim3 grid(M_TOT / 128, D_MODEL / 64, 3);
    qkv_gemm_kernel<<<grid, 256, 0, stream>>>(x, WqT, WkT, WvT, Qh, Kh, Vt);
  }
  {
    dim3 grid(SEQ / QB, BATCH);
    attn_kernel<<<grid, 256, 0, stream>>>(Qh, Kh, Vt, out);
  }
}
